// ModelSeedCorr_51488067944658
// MI455X (gfx1250) — matrix operand order repaired and checked by reading, not compiled, not run on hardware
//
#include <hip/hip_runtime.h>

// ---------------------------------------------------------------------------
// Radius-masked feature aggregation as a WMMA GEMM on gfx1250 (MI455X).
//   out[i,d] = sum_j [ ||seed_i - pt_j||^2 <= r ] * feats[j,d]
//
// Pass 1: feats f32[N][128] -> f16 transposed ws[128][N]  (halves L2 traffic)
// Pass 2: 64 seeds x 128 dims per block; K pipelined in 64-pt chunks with
//         double-buffered LDS. Feats chunks staged by the Tensor Data Mover
//         (tensor_load_to_lds, TENSORcnt) with LDS row padding applied by the
//         TDM descriptor; mask = exact f16 0/1; f32 accumulate via
//         v_wmma_f32_16x16x32_f16.
// ---------------------------------------------------------------------------

typedef _Float16 half_t;
typedef __attribute__((ext_vector_type(16))) _Float16 v16h;
typedef __attribute__((ext_vector_type(8)))  _Float16 v8h;
typedef __attribute__((ext_vector_type(8)))  float    v8f;
typedef __attribute__((ext_vector_type(4)))  unsigned int v4u;
typedef __attribute__((ext_vector_type(8)))  int          v8i;
typedef __attribute__((ext_vector_type(4)))  int          v4i;

#define N_PTS  32768
#define M_SUB  8192
#define D_FEAT 128
#define MTILE  64      // seeds per block
#define KC     64      // points (K) per staged chunk
#define LDSKP  72      // padded LDS row pitch in halves (144B; 16B-aligned,
                       // spreads the 64 LDS banks; matches TDM pad descriptor)

#define FT_SZ  (D_FEAT * LDSKP)   // halves per featsT buffer
#define MK_SZ  (MTILE * LDSKP)    // halves per mask buffer
#define PT_SZ  (KC * 3)           // floats per pts buffer

union Frag16 { v16h v; v8h h[2]; };

// ---------------------------------------------------------------------------
// Pass 1: f32 -> f16 transpose of the candidate feature matrix into d_ws.
// ws layout: [d][n], fixed row stride N_PTS halves.
// ---------------------------------------------------------------------------
__global__ __launch_bounds__(256)
void feats_transpose_f16(const float* __restrict__ enc_feats,
                         const float* __restrict__ sub_feats,
                         const int*   __restrict__ is_query,
                         unsigned short* __restrict__ ws_u)
{
    half_t* wsT = (half_t*)ws_u;
    const int q  = is_query[0];
    const float* src = q ? sub_feats : enc_feats;
    const int nk = q ? M_SUB : N_PTS;

    const int n0 = blockIdx.x * 64;
    if (n0 >= nk) return;               // uniform early-out (q==1 path)
    const int d0 = blockIdx.y * 64;

    __shared__ __align__(16) half_t tile[64 * LDSKP];

    const int tid = threadIdx.x;
    const int nl = tid >> 2;
    const int dl = (tid & 3) * 16;
    const float4* row =
        (const float4*)(src + (size_t)(n0 + nl) * D_FEAT + d0 + dl);
    #pragma unroll
    for (int i = 0; i < 4; ++i) {
        const float4 f = row[i];
        const int d = dl + i * 4;
        tile[(d + 0) * LDSKP + nl] = (half_t)f.x;
        tile[(d + 1) * LDSKP + nl] = (half_t)f.y;
        tile[(d + 2) * LDSKP + nl] = (half_t)f.z;
        tile[(d + 3) * LDSKP + nl] = (half_t)f.w;
    }
    __syncthreads();

    const int dw = tid >> 2;
    const int nw = (tid & 3) * 16;
    float4*       dst  = (float4*)(wsT + (size_t)(d0 + dw) * N_PTS + n0 + nw);
    const float4* srcT = (const float4*)&tile[dw * LDSKP + nw];
    dst[0] = srcT[0];
    dst[1] = srcT[1];
}

// ---------------------------------------------------------------------------
// Pass 2 helpers
// ---------------------------------------------------------------------------
__device__ __forceinline__ void stage_pts(const float* __restrict__ pts,
                                          int pb, float* s_pts, int tid)
{
    if (tid < PT_SZ / 4) {
        ((float4*)s_pts)[tid] = ((const float4*)(pts + (size_t)pb * 3))[tid];
    }
}

__device__ __forceinline__ void mask_compute(const float* s_pts, half_t* s_mask,
                                             int m_loc, int k_seg,
                                             float sx, float sy, float sz,
                                             float r)
{
    v8h m0, m1;
    #pragma unroll
    for (int k = 0; k < 16; ++k) {
        const int   kk = k_seg + k;
        const float dx = sx - s_pts[kk * 3 + 0];
        const float dy = sy - s_pts[kk * 3 + 1];
        const float dz = sz - s_pts[kk * 3 + 2];
        const float d2 = dx * dx + dy * dy + dz * dz;
        const half_t v = (d2 <= r) ? (half_t)1.0f : (half_t)0.0f;
        if (k < 8) m0[k] = v; else m1[k - 8] = v;
    }
    *(v8h*)&s_mask[m_loc * LDSKP + k_seg]     = m0;
    *(v8h*)&s_mask[m_loc * LDSKP + k_seg + 8] = m1;
}

// TDM descriptor + issue: copy 2D tile [rows=128 dims, cols=64 halves] from
// wsT (row stride N_PTS halves) into LDS at lds_byte, LDS pitch 144B via pad
// fields (32 DWORDs data then 4 DWORDs pad -> 128B + 16B = LDSKP*2).
__device__ __forceinline__ void tdm_stage_feats(const half_t* wsT, int pb,
                                                unsigned int lds_byte, int nk)
{
    const unsigned long long ga = (unsigned long long)(const void*)wsT +
                                  (unsigned long long)pb * 2ull;
    v4u g0;
    g0[0] = 1u;                                   // count=1, user mode
    g0[1] = lds_byte;                             // lds_addr
    g0[2] = (unsigned int)ga;                     // global_addr[31:0]
    g0[3] = (unsigned int)((ga >> 32) & 0x01FFFFFFull) | 0x80000000u; // type=2

    v8i g1;
    g1[0] = (int)((1u << 16) |                    // data_size = 2 bytes
                  (1u << 20) |                    // pad_enable
                  (4u << 22) |                    // pad_interval: 32 DWORDs
                  (3u << 25));                    // pad_amount:   4 DWORDs
    g1[1] = (int)(((unsigned)nk & 0xFFFFu) << 16);        // tensor_dim0 lo16
    g1[2] = (int)((((unsigned)nk >> 16) & 0xFFFFu) |      // tensor_dim0 hi16
                  ((unsigned)D_FEAT << 16));              // tensor_dim1 lo16
    g1[3] = (int)((unsigned)KC << 16);            // tensor_dim1 hi=0, tile_dim0=64
    g1[4] = (int)(unsigned)D_FEAT;                // tile_dim1=128, tile_dim2=0
    g1[5] = (int)(unsigned)N_PTS;                 // tensor_dim0_stride lo32
    g1[6] = 0;                                    // stride hi, dim1_stride lo
    g1[7] = 0;

    v4i g2 = {0, 0, 0, 0};                        // unused (2D tensor)
    v4i g3 = {0, 0, 0, 0};
    v8i g4 = {0, 0, 0, 0, 0, 0, 0, 0};            // 6-arg form extra operand

    // amdgpu-toolchain (clang-23) 6-arg form:
    // (uint32x4 g0, int32x8 g1, int32x4, int32x4, int32x8, i32 cpol)
    __builtin_amdgcn_tensor_load_to_lds(g0, g1, g2, g3, g4, 0);
}

// ---------------------------------------------------------------------------
// Pass 2: WMMA main kernel (double-buffered K pipeline, TDM staging)
// ---------------------------------------------------------------------------
__global__ __launch_bounds__(256)
void seedcorr_wmma_main(const float* __restrict__ enc_xyz,
                        const float* __restrict__ sub_xyz,
                        const float* __restrict__ crop_radius,
                        const int*   __restrict__ is_query,
                        const unsigned short* __restrict__ ws_u,
                        float*       __restrict__ out)
{
    const half_t* wsT = (const half_t*)ws_u;

    __shared__ __align__(16) half_t s_featsT[2 * FT_SZ]; // 36.0 KB
    __shared__ __align__(16) half_t s_mask  [2 * MK_SZ]; // 18.0 KB
    __shared__ __align__(16) float  s_pts   [2 * PT_SZ]; //  1.5 KB

    const int tid  = threadIdx.x;
    const int wave = tid >> 5;
    const int lane = tid & 31;
    const int col  = lane & 15;
    const int hi   = lane >> 4;

    const int m_idx   = wave & 3;       // 16-seed tile
    const int d_group = wave >> 2;      // 64-dim half

    const int q = is_query[0];
    const float* pts = q ? sub_xyz : enc_xyz;
    const int    nk  = q ? M_SUB   : N_PTS;
    const int nchunks = nk / KC;

    const float r = crop_radius[0];

    const int m_loc  = tid >> 2;
    const int k_seg  = (tid & 3) * 16;
    const int m_glob = blockIdx.x * MTILE + m_loc;
    const float sx = sub_xyz[m_glob * 3 + 0];
    const float sy = sub_xyz[m_glob * 3 + 1];
    const float sz = sub_xyz[m_glob * 3 + 2];

    // LDS byte base of each featsT buffer (low 32 bits of LDS aperture addr)
    const unsigned int ftlds0 =
        (unsigned int)(unsigned long long)(const void*)&s_featsT[0];
    const unsigned int ftlds1 =
        (unsigned int)(unsigned long long)(const void*)&s_featsT[FT_SZ];

    v8f acc[4] = {};

    // ---- prologue: stage + mask chunk 0 into buffer 0 ----
    stage_pts(pts, 0, s_pts, tid);
    if (wave == 0) tdm_stage_feats(wsT, 0, ftlds0, nk);
    __syncthreads();
    mask_compute(s_pts, s_mask, m_loc, k_seg, sx, sy, sz, r);
    if (wave == 0) __builtin_amdgcn_s_wait_tensorcnt(0);
    __syncthreads();

    for (int c = 0; c < nchunks; ++c) {
        const int cur = c & 1;
        const int nxt = cur ^ 1;
        const bool more = (c + 1) < nchunks;   // uniform

        // ---- issue next chunk's staging before this chunk's math ----
        if (more) {
            stage_pts(pts, (c + 1) * KC, s_pts + nxt * PT_SZ, tid);
            if (wave == 0)
                tdm_stage_feats(wsT, (c + 1) * KC, nxt ? ftlds1 : ftlds0, nk);
        }
        __syncthreads();   // pts[nxt] visible for mask computation

        if (more) {
            mask_compute(s_pts + nxt * PT_SZ, s_mask + nxt * MK_SZ,
                         m_loc, k_seg, sx, sy, sz, r);
        }

        // ---- WMMA on current buffer: batch all fragment loads first ----
        // A (16x32 f16): lane%16 = row M, K = hi*16 + h
        // B (32x16 f16): lane = column N, K = hi*16 + h
        const half_t* mk = s_mask   + cur * MK_SZ;
        const half_t* fT = s_featsT + cur * FT_SZ;

        Frag16 a[2];
        #pragma unroll
        for (int s = 0; s < 2; ++s) {
            const int arow = (m_idx * 16 + col) * LDSKP + s * 32;
            a[s].h[0] = *(const v8h*)&mk[arow + hi * 16];
            a[s].h[1] = *(const v8h*)&mk[arow + hi * 16 + 8];
        }
        Frag16 b[8];
        #pragma unroll
        for (int s = 0; s < 2; ++s) {
            #pragma unroll
            for (int t = 0; t < 4; ++t) {
                const int brow =
                    (d_group * 64 + t * 16 + col) * LDSKP + s * 32 + hi * 16;
                b[s * 4 + t].h[0] = *(const v8h*)&fT[brow];
                b[s * 4 + t].h[1] = *(const v8h*)&fT[brow + 8];
            }
        }
        #pragma unroll
        for (int s = 0; s < 2; ++s) {
            #pragma unroll
            for (int t = 0; t < 4; ++t) {
                acc[t] = __builtin_amdgcn_wmma_f32_16x16x32_f16(
                    false, a[s].v, false, b[s * 4 + t].v,
                    (short)0, acc[t], false, false);
            }
        }

        // next chunk's TDM data must be in LDS before it becomes 'cur'
        if (more && wave == 0) __builtin_amdgcn_s_wait_tensorcnt(0);
        __syncthreads();
    }

    // ---- store C tiles: VGPR r -> row (hi*8 + r), lane%16 -> column ----
    const int seed0 = blockIdx.x * MTILE + m_idx * 16 + hi * 8;
    #pragma unroll
    for (int t = 0; t < 4; ++t) {
        const int dcol = d_group * 64 + t * 16 + col;
        #pragma unroll
        for (int rr = 0; rr < 8; ++rr) {
            out[(size_t)(seed0 + rr) * D_FEAT + dcol] = acc[t][rr];
        }
    }
}

// ---------------------------------------------------------------------------
// Fallback (no workspace): single-pass kernel, converts in-loop.
// ---------------------------------------------------------------------------
__global__ __launch_bounds__(256)
void seedcorr_wmma_fallback(const float* __restrict__ enc_xyz,
                            const float* __restrict__ enc_feats,
                            const float* __restrict__ sub_xyz,
                            const float* __restrict__ sub_feats,
                            const float* __restrict__ crop_radius,
                            const int*   __restrict__ is_query,
                            float*       __restrict__ out)
{
    __shared__ __align__(16) half_t s_featsT[FT_SZ];
    __shared__ __align__(16) half_t s_mask  [MK_SZ];
    __shared__ __align__(16) float  s_pts   [PT_SZ];

    const int tid  = threadIdx.x;
    const int wave = tid >> 5;
    const int lane = tid & 31;
    const int col  = lane & 15;
    const int hi   = lane >> 4;
    const int m_idx   = wave & 3;
    const int d_group = wave >> 2;

    const int q = is_query[0];
    const float* pts   = q ? sub_xyz   : enc_xyz;
    const float* feats = q ? sub_feats : enc_feats;
    const int    nk    = q ? M_SUB     : N_PTS;
    const float r = crop_radius[0];

    const int m_loc  = tid >> 2;
    const int k_seg  = (tid & 3) * 16;
    const int m_glob = blockIdx.x * MTILE + m_loc;
    const float sx = sub_xyz[m_glob * 3 + 0];
    const float sy = sub_xyz[m_glob * 3 + 1];
    const float sz = sub_xyz[m_glob * 3 + 2];

    const int f_k  = tid >> 2;
    const int f_d4 = (tid & 3) * 8;

    v8f acc[4] = {};

    for (int pb = 0; pb < nk; pb += KC) {
        stage_pts(pts, pb, s_pts, tid);
        __syncthreads();
        const float4* frow =
            (const float4*)(feats + (size_t)(pb + f_k) * D_FEAT);
        #pragma unroll
        for (int i = 0; i < 8; ++i) {
            const int    d4 = f_d4 + i;
            const float4 f  = frow[d4];
            const int    d  = d4 * 4;
            s_featsT[(d + 0) * LDSKP + f_k] = (half_t)f.x;
            s_featsT[(d + 1) * LDSKP + f_k] = (half_t)f.y;
            s_featsT[(d + 2) * LDSKP + f_k] = (half_t)f.z;
            s_featsT[(d + 3) * LDSKP + f_k] = (half_t)f.w;
        }
        mask_compute(s_pts, s_mask, m_loc, k_seg, sx, sy, sz, r);
        __syncthreads();
        #pragma unroll
        for (int kk = 0; kk < KC; kk += 32) {
            Frag16 a;
            const int arow = (m_idx * 16 + col) * LDSKP + kk;
            a.h[0] = *(const v8h*)&s_mask[arow + hi * 16];
            a.h[1] = *(const v8h*)&s_mask[arow + hi * 16 + 8];
            #pragma unroll
            for (int t = 0; t < 4; ++t) {
                Frag16 b;
                const int brow =
                    (d_group * 64 + t * 16 + col) * LDSKP + kk + hi * 16;
                b.h[0] = *(const v8h*)&s_featsT[brow];
                b.h[1] = *(const v8h*)&s_featsT[brow + 8];
                acc[t] = __builtin_amdgcn_wmma_f32_16x16x32_f16(
                    false, a.v, false, b.v, (short)0, acc[t], false, false);
            }
        }
        __syncthreads();
    }

    const int seed0 = blockIdx.x * MTILE + m_idx * 16 + hi * 8;
    #pragma unroll
    for (int t = 0; t < 4; ++t) {
        const int dcol = d_group * 64 + t * 16 + col;
        #pragma unroll
        for (int rr = 0; rr < 8; ++rr) {
            out[(size_t)(seed0 + rr) * D_FEAT + dcol] = acc[t][rr];
        }
    }
}

// ---------------------------------------------------------------------------
extern "C" void kernel_launch(void* const* d_in, const int* in_sizes, int n_in,
                              void* d_out, int out_size, void* d_ws, size_t ws_size,
                              hipStream_t stream) {
    (void)in_sizes; (void)n_in; (void)out_size;
    const float* enc_xyz   = (const float*)d_in[0];
    const float* enc_feats = (const float*)d_in[1];
    const float* sub_xyz   = (const float*)d_in[2];
    const float* sub_feats = (const float*)d_in[3];
    const float* radius    = (const float*)d_in[4];
    const int*   is_query  = (const int*)d_in[5];
    float* out = (float*)d_out;

    const size_t ws_need = (size_t)D_FEAT * N_PTS * 2u;  // 8 MB f16 transpose

    if (ws_size >= ws_need && d_ws != nullptr) {
        unsigned short* wsT = (unsigned short*)d_ws;
        hipLaunchKernelGGL(feats_transpose_f16,
                           dim3(N_PTS / 64, D_FEAT / 64), dim3(256), 0, stream,
                           enc_feats, sub_feats, is_query, wsT);
        hipLaunchKernelGGL(seedcorr_wmma_main,
                           dim3(M_SUB / MTILE), dim3(256), 0, stream,
                           enc_xyz, sub_xyz, radius, is_query, wsT, out);
    } else {
        hipLaunchKernelGGL(seedcorr_wmma_fallback,
                           dim3(M_SUB / MTILE), dim3(256), 0, stream,
                           enc_xyz, enc_feats, sub_xyz, sub_feats,
                           radius, is_query, out);
    }
}
